// BasicBlock_91311004713001
// MI455X (gfx1250) — matrix operand order repaired and checked by reading, not compiled, not run on hardware
//
#include <hip/hip_runtime.h>
#include <stdint.h>

typedef __attribute__((ext_vector_type(8)))  float          v8f;
typedef __attribute__((ext_vector_type(16))) __bf16         v16bf;
typedef __attribute__((ext_vector_type(16))) unsigned short v16u;

union FragU {
    v16u           u;
    v16bf          b;
    uint4          q[2];
    unsigned short h[16];
    unsigned int   d[8];
};

// ---------------------------------------------------------------------------
// Branchless closed-form of the 128 sequential torch.where posit passes.
// grid points (1 + j/8)*2^E;  m = mantissa in [1,2), jlo = top 3 mantissa
// bits, low = next 20 bits (mid = 0x80000 marks the interval boundaries
// m = 1.0625 + k/8).
//   tiny:  0 < |x| < 1.0625*2^-16            -> +-2^-16
//   |x|<1: m in (1.0625,1.9375), not boundary -> round to grid, else pass
//   |x|>=1:m in [1,1.0625)                   -> 2^E
//          m in (1.875,2) (incl m=1.9375!)   -> 2^(E+1)   (E<15; later
//          torch pass (E+1,j=0) covers it and wins)
//          other non-boundary m              -> round to grid
//          boundaries / E>15 / E==15 tail    -> pass
// All conditions are flat compares -> v_cmp + cndmask, no divergence.
// ---------------------------------------------------------------------------
__device__ __forceinline__ float posit_q(float x) {
    unsigned int u  = __float_as_uint(x);
    unsigned int au = u & 0x7fffffffu;
    unsigned int sg = u & 0x80000000u;
    int E = (int)(au >> 23) - 127;
    unsigned int mant = au & 0x007fffffu;
    unsigned int jlo  = mant >> 20;                 // 0..7
    unsigned int low  = mant & 0x000fffffu;
    bool bnd  = (low == 0x00080000u);               // m = 1.0625 + k/8 exactly
    unsigned int j = jlo + ((low > 0x00080000u) ? 1u : 0u);  // rounded grid idx

    bool tiny = (au != 0u) && (au < 0x37880000u);   // |x| < 1.0625*2^-16
    bool up   = (E >= 0) && (E < 15) && (jlo == 7u) && (low != 0u);
    bool pass = (au == 0u) || (E > 15)
             || ((E < 0)  && (bnd || (j - 1u > 6u)))
             || ((E >= 0) && bnd && (jlo != 7u))
             || ((E == 15) && (jlo == 7u) && (low >= 0x00080000u));

    unsigned int r_up   = sg | ((unsigned int)(E + 128) << 23);
    unsigned int r_grid = sg | ((unsigned int)(E + 127) << 23) | (j << 20);
    unsigned int r = up ? r_up : r_grid;
    r = pass ? u : r;
    r = tiny ? (sg | 0x37800000u) : r;
    return __uint_as_float(r);
}

// f32 -> bf16, round-to-nearest-even (exact for quantized grid values)
__device__ __forceinline__ unsigned short f2bf(float f) {
    unsigned int u = __float_as_uint(f);
    u += 0x7fffu + ((u >> 16) & 1u);
    return (unsigned short)(u >> 16);
}

#define C_CH      256
#define DHW       16384
#define W1_OFF    0
#define W2_OFF    131072
#define AB_OFF    262144
#define LDS_BYTES (262144 + 4096)
#define YSTR      264          // bf16 halves per spatial row (256 + 8 pad)

// ---------------------------------------------------------------------------
// Pre-kernel: quantize W1/W2 to bf16 in WMMA A-fragment order; fold
// conv-bias + BN into per-channel (scale, shift).
// Fragment (kt, mt): 32 lanes x 16 halves contiguous.  Lane L (L<16: M-row L,
// K 0..15; L>=16: K 16..31), contiguous k = 16*(L/16) + h, the same
// order the B fragments use.
// ---------------------------------------------------------------------------
__global__ void bb_prep_kernel(const float* __restrict__ w1, const float* __restrict__ w2,
                               const float* __restrict__ b1, const float* __restrict__ g1,
                               const float* __restrict__ be1, const float* __restrict__ m1,
                               const float* __restrict__ v1,
                               const float* __restrict__ b2, const float* __restrict__ g2,
                               const float* __restrict__ be2, const float* __restrict__ m2,
                               const float* __restrict__ v2,
                               unsigned char* __restrict__ ws) {
    int t    = blockIdx.x * 256 + threadIdx.x;   // 0 .. 16383
    int mat  = t >> 13;
    int r    = t & 8191;
    int frag = r >> 5;                           // 0..255 : kt*16 + mt
    int lane = r & 31;
    int kt   = frag >> 4, mt = frag & 15;
    const float* w = mat ? w2 : w1;
    int M    = (mt << 4) | (lane & 15);
    int run0 = (lane & 16) >> 1;                 // 0 or 8
    float raw[16];
#pragma unroll
    for (int h = 0; h < 16; ++h) {
        int K = (lane & 16) + h;                 // {16*hi .. 16*hi+15}
        raw[h] = w[M * 256 + (kt << 5) + K];
    }
    unsigned int o[8];
#pragma unroll
    for (int h = 0; h < 16; ++h) {
        unsigned short hb = f2bf(posit_q(raw[h]));
        if (h & 1) o[h >> 1] |= ((unsigned int)hb) << 16;
        else       o[h >> 1]  = hb;
    }
    uint4* dst = (uint4*)(ws + (size_t)mat * 131072 + (size_t)(frag * 32 + lane) * 32);
    dst[0] = make_uint4(o[0], o[1], o[2], o[3]);
    dst[1] = make_uint4(o[4], o[5], o[6], o[7]);

    if (t < 512) {   // BN folds: y*scale + (bias*scale + beta - mean*scale)
        int c = t & 255;
        const float* bb  = (t < 256) ? b1  : b2;
        const float* gg  = (t < 256) ? g1  : g2;
        const float* bee = (t < 256) ? be1 : be2;
        const float* mm  = (t < 256) ? m1  : m2;
        const float* vv  = (t < 256) ? v1  : v2;
        float s = gg[c] * rsqrtf(vv[c] + 1e-5f);
        float2 p;
        p.x = s;
        p.y = bb[c] * s + bee[c] - mm[c] * s;
        *(float2*)(ws + AB_OFF + (size_t)t * 8) = p;
    }
}

// ---------------------------------------------------------------------------
// Fused kernel: quant -> GEMM1(bf16 WMMA) -> BN1 -> relu -> quant ->
//               GEMM2(bf16 WMMA) -> BN2 -> +residual -> relu
// 1024 blocks x 256 threads; block = (batch n, 128 spatial).
// Wave split: mh = wave>>2 selects channels mh*128..+127 (8 M-tiles),
// cpair = wave&3 selects 2 column-tiles (32 spatial).  Each A-fragment
// (ds_load_b128 x2) feeds TWO WMMAs.  B-build: 32 batched global loads,
// then branchless quantize -> no per-element load/wait serialization.
// ---------------------------------------------------------------------------
__global__ __launch_bounds__(256) void bb_fused_kernel(const float* __restrict__ x,
                                                       const unsigned short* __restrict__ wfrag,
                                                       const float2* __restrict__ ab,
                                                       float* __restrict__ out) {
    extern __shared__ char lds[];
    unsigned tid = threadIdx.x;

    // Stage W1 + W2 fragments (256 KB) into LDS, coalesced 16B chunks.
    {
        const uint4* src = (const uint4*)wfrag;
        uint4* dst = (uint4*)lds;
#pragma unroll 4
        for (int i = 0; i < 64; ++i) dst[tid + 256 * i] = src[tid + 256 * i];
        float2* dab = (float2*)(lds + AB_OFF);
        dab[tid]       = ab[tid];
        dab[tid + 256] = ab[tid + 256];
    }
    __syncthreads();

    const int n     = blockIdx.x >> 7;
    const int s0    = (blockIdx.x & 127) << 7;
    const int wave  = tid >> 5;
    const int lane  = tid & 31;
    const int lr    = lane & 15;
    const int hi    = lane >> 4;             // 0/1: K-half selector
    const int mh    = wave >> 2;             // M half: channels mh*128..+127
    const int cp    = wave & 3;              // column pair: tiles 2cp, 2cp+1
    const int sl0   = ((cp << 1) | 0) * 16 + lr;   // column 0 (0..127)
    const int sl1   = ((cp << 1) | 1) * 16 + lr;   // column 1
    const size_t sg0 = (size_t)(s0 + sl0);
    const size_t sg1 = (size_t)(s0 + sl1);

    const float* xn = x + (size_t)n * C_CH * DHW;
    const float2* dab = (const float2*)(lds + AB_OFF);

    // ---------------- GEMM1: acc[mt][ct] = Wq1 x Xq ----------------
    v8f acc[8][2];
#pragma unroll
    for (int mt = 0; mt < 8; ++mt) {
        acc[mt][0] = (v8f){0.f,0.f,0.f,0.f,0.f,0.f,0.f,0.f};
        acc[mt][1] = (v8f){0.f,0.f,0.f,0.f,0.f,0.f,0.f,0.f};
    }

    for (int kt = 0; kt < 8; ++kt) {
        const int cb = (kt << 5) + (hi << 4);
        const float* px0 = xn + (size_t)cb * DHW + sg0;
        const float* px1 = xn + (size_t)cb * DHW + sg1;
        if (kt < 7) __builtin_prefetch(px0 + 32 * DHW, 0, 1);

        float raw0[16], raw1[16];            // phase 1: batched loads
#pragma unroll
        for (int i = 0; i < 16; ++i) {
            raw0[i] = px0[(size_t)i * DHW];
            raw1[i] = px1[(size_t)i * DHW];
        }
        FragU B0, B1;                        // phase 2: branchless quantize
#pragma unroll
        for (int i = 0; i < 16; ++i) {
            B0.h[i] = f2bf(posit_q(raw0[i]));
            B1.h[i] = f2bf(posit_q(raw1[i]));
        }

        FragU A[2];
        {
            const uint4* ap = (const uint4*)(lds + W1_OFF +
                               (size_t)(((kt << 4) | (mh << 3)) * 32 + lane) * 32);
            A[0].q[0] = ap[0];
            A[0].q[1] = ap[1];
        }
#pragma unroll
        for (int mt = 0; mt < 8; ++mt) {
            if (mt < 7) {       // issue next fragment's loads before these WMMAs
                const uint4* ap = (const uint4*)(lds + W1_OFF +
                                   (size_t)(((kt << 4) | (mh << 3) | (mt + 1)) * 32 + lane) * 32);
                A[(mt + 1) & 1].q[0] = ap[0];
                A[(mt + 1) & 1].q[1] = ap[1];
            }
            acc[mt][0] = __builtin_amdgcn_wmma_f32_16x16x32_bf16(
                false, A[mt & 1].b, false, B0.b, (short)0, acc[mt][0], false, false);
            acc[mt][1] = __builtin_amdgcn_wmma_f32_16x16x32_bf16(
                false, A[mt & 1].b, false, B1.b, (short)0, acc[mt][1], false, false);
        }
    }

    __syncthreads();   // all waves done reading W1 region before Y1 aliases it

    // ---------------- BN1 + relu + quantize -> Y1 (bf16, LDS) -------------
    unsigned short* y1 = (unsigned short*)(lds + W1_OFF);
#pragma unroll
    for (int mt = 0; mt < 8; ++mt) {
        const int c0 = (mh << 7) | (mt << 4) | (hi << 3);   // 8 contiguous ch
        unsigned int o4a[4], o4b[4];
#pragma unroll
        for (int rr = 0; rr < 8; ++rr) {
            float2 p = dab[c0 + rr];
            float va = fmaxf(acc[mt][0][rr] * p.x + p.y, 0.f);
            float vb = fmaxf(acc[mt][1][rr] * p.x + p.y, 0.f);
            unsigned short ha = f2bf(posit_q(va));
            unsigned short hb = f2bf(posit_q(vb));
            if (rr & 1) { o4a[rr >> 1] |= ((unsigned int)ha) << 16;
                          o4b[rr >> 1] |= ((unsigned int)hb) << 16; }
            else        { o4a[rr >> 1]  = ha;
                          o4b[rr >> 1]  = hb; }
        }
        *(uint4*)(y1 + (size_t)sl0 * YSTR + c0) = make_uint4(o4a[0], o4a[1], o4a[2], o4a[3]);
        *(uint4*)(y1 + (size_t)sl1 * YSTR + c0) = make_uint4(o4b[0], o4b[1], o4b[2], o4b[3]);
    }

    __syncthreads();   // Y1 columns are filled by BOTH M-half wave groups

    // ---------------- GEMM2: acc2[mt][ct] = Wq2 x Y1q ----------------
    v8f acc2[8][2];
#pragma unroll
    for (int mt = 0; mt < 8; ++mt) {
        acc2[mt][0] = (v8f){0.f,0.f,0.f,0.f,0.f,0.f,0.f,0.f};
        acc2[mt][1] = (v8f){0.f,0.f,0.f,0.f,0.f,0.f,0.f,0.f};
    }

    for (int kt = 0; kt < 8; ++kt) {
        FragU B0, B1;
        {
            const uint4* bp0 = (const uint4*)(y1 + (size_t)sl0 * YSTR + (kt << 5) + (hi << 4));
            const uint4* bp1 = (const uint4*)(y1 + (size_t)sl1 * YSTR + (kt << 5) + (hi << 4));
            B0.q[0] = bp0[0];
            B0.q[1] = bp0[1];
            B1.q[0] = bp1[0];
            B1.q[1] = bp1[1];
        }
        FragU A[2];
        {
            const uint4* ap = (const uint4*)(lds + W2_OFF +
                               (size_t)(((kt << 4) | (mh << 3)) * 32 + lane) * 32);
            A[0].q[0] = ap[0];
            A[0].q[1] = ap[1];
        }
#pragma unroll
        for (int mt = 0; mt < 8; ++mt) {
            if (mt < 7) {
                const uint4* ap = (const uint4*)(lds + W2_OFF +
                                   (size_t)(((kt << 4) | (mh << 3) | (mt + 1)) * 32 + lane) * 32);
                A[(mt + 1) & 1].q[0] = ap[0];
                A[(mt + 1) & 1].q[1] = ap[1];
            }
            acc2[mt][0] = __builtin_amdgcn_wmma_f32_16x16x32_bf16(
                false, A[mt & 1].b, false, B0.b, (short)0, acc2[mt][0], false, false);
            acc2[mt][1] = __builtin_amdgcn_wmma_f32_16x16x32_bf16(
                false, A[mt & 1].b, false, B1.b, (short)0, acc2[mt][1], false, false);
        }
    }

    // ---------------- BN2 + residual + relu -> out ----------------
    float* outn = out + (size_t)n * C_CH * DHW;
    const float2* dab2 = dab + 256;
#pragma unroll
    for (int mt = 0; mt < 8; ++mt) {
        const int c0 = (mh << 7) | (mt << 4) | (hi << 3);
#pragma unroll
        for (int rr = 0; rr < 8; ++rr) {
            const int c = c0 + rr;
            float2 p = dab2[c];
            float r0 = xn[(size_t)c * DHW + sg0];
            float r1 = xn[(size_t)c * DHW + sg1];
            float v0 = acc2[mt][0][rr] * p.x + p.y + r0;
            float v1 = acc2[mt][1][rr] * p.x + p.y + r1;
            outn[(size_t)c * DHW + sg0] = fmaxf(v0, 0.f);
            outn[(size_t)c * DHW + sg1] = fmaxf(v1, 0.f);
        }
    }
}

extern "C" void kernel_launch(void* const* d_in, const int* in_sizes, int n_in,
                              void* d_out, int out_size, void* d_ws, size_t ws_size,
                              hipStream_t stream) {
    const float* x   = (const float*)d_in[0];
    const float* w1  = (const float*)d_in[1];
    const float* b1  = (const float*)d_in[2];
    const float* g1  = (const float*)d_in[3];
    const float* be1 = (const float*)d_in[4];
    const float* m1  = (const float*)d_in[5];
    const float* v1  = (const float*)d_in[6];
    const float* w2  = (const float*)d_in[7];
    const float* b2  = (const float*)d_in[8];
    const float* g2  = (const float*)d_in[9];
    const float* be2 = (const float*)d_in[10];
    const float* m2  = (const float*)d_in[11];
    const float* v2  = (const float*)d_in[12];
    unsigned char* ws = (unsigned char*)d_ws;
    float* out = (float*)d_out;

    bb_prep_kernel<<<64, 256, 0, stream>>>(w1, w2, b1, g1, be1, m1, v1,
                                           b2, g2, be2, m2, v2, ws);

    (void)hipFuncSetAttribute((const void*)bb_fused_kernel,
                              hipFuncAttributeMaxDynamicSharedMemorySize, LDS_BYTES);
    bb_fused_kernel<<<1024, 256, LDS_BYTES, stream>>>(
        x, (const unsigned short*)ws, (const float2*)(ws + AB_OFF), out);
}
